// FlashMultiheadAttention_56925496541188
// MI455X (gfx1250) — matrix operand order repaired and checked by reading, not compiled, not run on hardware
//
#include <hip/hip_runtime.h>
#include <hip/hip_bf16.h>

typedef _Float16 half_t;
typedef __attribute__((ext_vector_type(16))) _Float16 v16h;
typedef __attribute__((ext_vector_type(8)))  _Float16 h8;
typedef __attribute__((ext_vector_type(8)))  float    v8f;
typedef __attribute__((ext_vector_type(4)))  float    f4;
typedef __attribute__((ext_vector_type(4)))  unsigned int v4u;
typedef __attribute__((ext_vector_type(8)))  int          v8i;
typedef __attribute__((ext_vector_type(4)))  int          v4i;

#define SEQ_   2048
#define DMODEL 2048
#define HEADS  16
#define HD     128

static __device__ __forceinline__ v8f wmma_f16(v16h a, v16h b, v8f c) {
  // D = A(16x32 f16) * B(32x16 f16) + C(16x16 f32)
  return __builtin_amdgcn_wmma_f32_16x16x32_f16(
      /*neg_a=*/false, a, /*neg_b=*/false, b,
      /*c_mod=*/(short)0, c, /*reuse_a=*/false, /*reuse_b=*/false);
}

static __device__ __forceinline__ v16h ldfrag(const half_t* p0, const half_t* p1) {
  v16h r;
  ((h8*)&r)[0] = *(const h8*)p0;
  ((h8*)&r)[1] = *(const h8*)p1;
  return r;
}

static __device__ __forceinline__ v8f zero8() {
  v8f z;
#pragma unroll
  for (int i = 0; i < 8; ++i) z[i] = 0.0f;
  return z;
}

// Issue a TDM 2D tile load: rows x 128 f16 elements, global row stride 2048
// elements, into LDS at lds_byte_addr.  D# layout per CDNA5 ISA ch.8.
// clang-23 toolchain: 6-arg builtin (g0 v4u, g1 v8i, g2 v4i, g3 v4i, v8i, cpol).
static __device__ __forceinline__ void tdm_load_tile_f16(
    unsigned int lds_byte_addr, const half_t* gptr, int rows) {
  const unsigned long long ga = (unsigned long long)(uintptr_t)gptr;
  v4u g0;
  g0.x = 1u;                                  // count=1, user descriptor
  g0.y = lds_byte_addr;                       // lds_addr[31:0]
  g0.z = (unsigned int)ga;                    // global_addr[31:0]
  g0.w = ((unsigned int)(ga >> 32) & 0x01FFFFFFu) | (2u << 30); // addr[56:32] | type=2
  v8i g1;
  g1[0] = (1 << 16);                          // workgroup_mask=0, data_size=1 (2B)
  g1[1] = (DMODEL << 16);                     // tensor_dim0[15:0]=2048 (bits 63:48)
  g1[2] = (SEQ_ << 16);                       // dim0 hi=0 | tensor_dim1[15:0]=2048
  g1[3] = (HD << 16);                         // dim1 hi=0 | tile_dim0=128
  g1[4] = rows;                               // tile_dim1=rows, tile_dim2=0
  g1[5] = DMODEL;                             // tensor_dim0_stride[31:0]=2048 elems
  g1[6] = 0;
  g1[7] = 0;
  v4i gz4;
  gz4[0] = 0; gz4[1] = 0; gz4[2] = 0; gz4[3] = 0; // 2D tile: groups 2/3 unused
  v8i gz8;
#pragma unroll
  for (int i = 0; i < 8; ++i) gz8[i] = 0;
  __builtin_amdgcn_tensor_load_to_lds(g0, g1, gz4, gz4, gz8, 0);
}

// ---------------------------------------------------------------------------
// C[M,N] = A[M,K] @ B[N,K]^T  (A,B f32 in HBM, converted to f16 while staging
// into double-buffered LDS; f32 accumulation via v_wmma_f32_16x16x32_f16).
// Block tile 128x128, 8 waves, wave tile 64(M) x 32(N) = 4x2 WMMA tiles.
// Pipeline: fetch tile k+1 into registers while WMMAs consume LDS buffer k.
// ---------------------------------------------------------------------------
template <typename OutT>
__global__ __launch_bounds__(256)
void gemm_nt_kernel(const float* __restrict__ A, const float* __restrict__ B,
                    OutT* __restrict__ C, int M, int N, int K) {
  __shared__ half_t As[2][128 * 32];
  __shared__ half_t Bs[2][128 * 32];

  const int tid  = threadIdx.x;
  const int wid  = tid >> 5;        // 0..7
  const int lane = tid & 31;
  const int lh   = lane & 15;
  const int hi   = lane >> 4;       // 0|1 : lane-half
  const int bm   = blockIdx.y * 128;
  const int bn   = blockIdx.x * 128;
  const int wm   = (wid & 1) * 64;  // wave M offset inside block tile
  const int wn   = (wid >> 1) * 32; // wave N offset inside block tile

  const int lrow = tid >> 1;        // 0..127 (staging row)
  const int lc16 = (tid & 1) * 16;  // 0|16   (k sub-offset)
  const float* aptr = A + (size_t)(bm + lrow) * K + lc16;
  const float* bptr = B + (size_t)(bn + lrow) * K + lc16;

  f4 ar[4], br[4];
  auto fetch = [&](int k0) {
#pragma unroll
    for (int i = 0; i < 4; ++i) {
      ar[i] = *(const f4*)(aptr + k0 + i * 4);
      br[i] = *(const f4*)(bptr + k0 + i * 4);
    }
    __builtin_prefetch(aptr + k0 + 32, 0, 1);
    __builtin_prefetch(bptr + k0 + 32, 0, 1);
  };
  auto stage = [&](int buf) {
    h8 a0, a1, b0, b1;
#pragma unroll
    for (int i = 0; i < 4; ++i) {
      a0[i] = (half_t)ar[0][i]; a0[4 + i] = (half_t)ar[1][i];
      a1[i] = (half_t)ar[2][i]; a1[4 + i] = (half_t)ar[3][i];
      b0[i] = (half_t)br[0][i]; b0[4 + i] = (half_t)br[1][i];
      b1[i] = (half_t)br[2][i]; b1[4 + i] = (half_t)br[3][i];
    }
    *(h8*)(&As[buf][lrow * 32 + lc16])     = a0;
    *(h8*)(&As[buf][lrow * 32 + lc16 + 8]) = a1;
    *(h8*)(&Bs[buf][lrow * 32 + lc16])     = b0;
    *(h8*)(&Bs[buf][lrow * 32 + lc16 + 8]) = b1;
  };

  v8f acc[4][2];
#pragma unroll
  for (int mt = 0; mt < 4; ++mt)
#pragma unroll
    for (int nt = 0; nt < 2; ++nt) acc[mt][nt] = zero8();

  fetch(0);
  stage(0);
  __syncthreads();

  for (int k0 = 0; k0 < K; k0 += 32) {
    const int cur = (k0 >> 5) & 1;
    const bool more = (k0 + 32) < K;
    if (more) fetch(k0 + 32);   // overlap next-tile global loads with compute

    v16h af[4], bf[2];
    const int ko = hi * 8;   // A layout: K {0..7,16..23} | {8..15,24..31}
    const int kb = hi * 8;   // B layout: K {0..7,16..23} | {8..15,24..31}
#pragma unroll
    for (int mt = 0; mt < 4; ++mt) {
      const half_t* p = &As[cur][(wm + mt * 16 + lh) * 32];
      af[mt] = ldfrag(p + ko, p + ko + 16);
    }
#pragma unroll
    for (int nt = 0; nt < 2; ++nt) {
      const half_t* p = &Bs[cur][(wn + nt * 16 + lh) * 32];
      bf[nt] = ldfrag(p + kb, p + kb + 16);
    }
#pragma unroll
    for (int mt = 0; mt < 4; ++mt)
#pragma unroll
      for (int nt = 0; nt < 2; ++nt)
        acc[mt][nt] = wmma_f16(af[mt], bf[nt], acc[mt][nt]);

    if (more) {
      stage(cur ^ 1);
      __syncthreads();
    }
  }

  // ---- store (C/D layout: VGPR r -> M = r + 8*hi, N = lane&15) ----
#pragma unroll
  for (int mt = 0; mt < 4; ++mt)
#pragma unroll
    for (int nt = 0; nt < 2; ++nt)
#pragma unroll
      for (int r = 0; r < 8; ++r) {
        const int row = bm + wm + mt * 16 + r + hi * 8;
        const int col = bn + wn + nt * 16 + lh;
        C[(size_t)row * N + col] = (OutT)acc[mt][nt][r];
      }
}

// ---------------------------------------------------------------------------
// In-place rotate-half RoPE on f16 (S, H, hd) data; one thread per (s,h,pair).
// ---------------------------------------------------------------------------
__global__ __launch_bounds__(256)
void rope_kernel(half_t* __restrict__ X) {
  const int idx = blockIdx.x * 256 + threadIdx.x; // S*H*64 total
  const int i  = idx & 63;
  const int sh = idx >> 6;
  const int h  = sh & (HEADS - 1);
  const int s  = sh >> 4;
  const size_t base = (size_t)s * DMODEL + h * HD + i;
  const float x1 = (float)X[base];
  const float x2 = (float)X[base + 64];
  // freq = theta^(-2i/hd) = exp(-i * ln(10000)/64)
  const float freq  = __expf(-0.14391156831212787f * (float)i);
  const float angle = (float)s * freq;
  float sn, cs;
  __sincosf(angle, &sn, &cs);
  X[base]      = (half_t)(x1 * cs - x2 * sn);
  X[base + 64] = (half_t)(x2 * cs + x1 * sn);
}

// ---------------------------------------------------------------------------
// Flash attention, causal, one head x 64 query rows per block (4 waves).
// Each wave: 16 rows, full hd=128 in registers (8 f32 accum tiles).
// K tile staged into LDS by the Tensor Data Mover (one descriptor per wave,
// TENSORcnt-tracked); V tile staged transposed through VGPRs.  All WMMA
// fragments except Q (loaded from global) come from ds_load_b128.
// ---------------------------------------------------------------------------
__global__ __launch_bounds__(128)
void fa_kernel(const half_t* __restrict__ Qh, const half_t* __restrict__ Kh,
               const half_t* __restrict__ Vh, float* __restrict__ Of) {
  __shared__ half_t Ks[32 * HD];       // K tile row-major [j][d], 8 KB (TDM dest)
  __shared__ half_t Vt[HD * 32];       // V tile transposed [d][j], 8 KB
  __shared__ half_t Ps[4 * 16 * 32];   // per-wave P scratch (C->A relayout)

  const int tid  = threadIdx.x;
  const int wid  = tid >> 5;
  const int lane = tid & 31;
  const int lh   = lane & 15;
  const int hi   = lane >> 4;
  const int h    = blockIdx.y;
  const int qb   = blockIdx.x * 64;
  const int qrow0 = qb + wid * 16;

  const unsigned int ksBase =
      (unsigned int)(uintptr_t)(void*)&Ks[0] + (unsigned int)(wid * 8 * HD * 2);

  // ---- Q fragments (A layout), 4 K-steps covering hd=128 ----
  v16h qf[4];
  {
    const half_t* qp = Qh + (size_t)(qrow0 + lh) * DMODEL + h * HD;
#pragma unroll
    for (int ks = 0; ks < 4; ++ks) {
      const int ko = ks * 32 + hi * 8;
      qf[ks] = ldfrag(qp + ko, qp + ko + 16);
    }
  }

  v8f o[8];
  float m[8], l[8];
#pragma unroll
  for (int r = 0; r < 8; ++r) { m[r] = -1e30f; l[r] = 0.0f; }
#pragma unroll
  for (int dt = 0; dt < 8; ++dt) o[dt] = zero8();

  const float scale = 0.08838834764831845f; // 1/sqrt(128)
  const int nJT = (qb + 64) >> 5;           // causal: j tiles of 32

  for (int jt = 0; jt < nJT; ++jt) {
    const int j0 = jt * 32;

    // ---- TDM: each wave DMAs its 8-row quarter of the K tile to LDS ----
    tdm_load_tile_f16(ksBase, Kh + (size_t)(j0 + wid * 8) * DMODEL + h * HD, 8);

    // ---- cooperative staging of V tile (transposed) through VGPRs ----
    {
      const int r  = tid >> 2;         // 0..31 : j within tile
      const int db = (tid & 3) * 32;   // d sub-block
      const half_t* vp = Vh + (size_t)(j0 + r) * DMODEL + h * HD + db;
      h8 v0 = *(const h8*)(vp + 0);
      h8 v1 = *(const h8*)(vp + 8);
      h8 v2 = *(const h8*)(vp + 16);
      h8 v3 = *(const h8*)(vp + 24);
#pragma unroll
      for (int i = 0; i < 8; ++i) {
        Vt[(db + i) * 32 + r]      = v0[i];
        Vt[(db + 8 + i) * 32 + r]  = v1[i];
        Vt[(db + 16 + i) * 32 + r] = v2[i];
        Vt[(db + 24 + i) * 32 + r] = v3[i];
      }
    }
    __builtin_amdgcn_s_wait_tensorcnt(0);   // K tile DMA complete
    __syncthreads();

    // ---- S = Q @ K^T (two 16x16 tiles across j) ----
    v8f s[2];
#pragma unroll
    for (int nt = 0; nt < 2; ++nt) {
      const half_t* kp = Ks + (nt * 16 + lh) * HD;
      v16h kf[4];
#pragma unroll
      for (int ks = 0; ks < 4; ++ks) {
        const int kb = ks * 32 + hi * 8; // B layout: K {0..7,16..23} | {8..15,24..31}
        kf[ks] = ldfrag(kp + kb, kp + kb + 16);
      }
      v8f a = zero8();
#pragma unroll
      for (int ks = 0; ks < 4; ++ks) a = wmma_f16(qf[ks], kf[ks], a);
      s[nt] = a;
    }

    // ---- scale + causal mask + online softmax (fully unrolled) ----
    float p0[8], p1[8], alpha[8];
#pragma unroll
    for (int r = 0; r < 8; ++r) {
      const int irow = qrow0 + r + hi * 8;
      float v0 = s[0][r] * scale;
      float v1 = s[1][r] * scale;
      if (j0 + lh > irow)      v0 = -1e30f;
      if (j0 + 16 + lh > irow) v1 = -1e30f;
      float mx = fmaxf(v0, v1);
      mx = fmaxf(mx, __shfl_xor(mx, 1, 32));
      mx = fmaxf(mx, __shfl_xor(mx, 2, 32));
      mx = fmaxf(mx, __shfl_xor(mx, 4, 32));
      mx = fmaxf(mx, __shfl_xor(mx, 8, 32));
      const float mn = fmaxf(m[r], mx);
      alpha[r] = __expf(m[r] - mn);
      const float e0 = __expf(v0 - mn);
      const float e1 = __expf(v1 - mn);
      float rs = e0 + e1;
      rs += __shfl_xor(rs, 1, 32);
      rs += __shfl_xor(rs, 2, 32);
      rs += __shfl_xor(rs, 4, 32);
      rs += __shfl_xor(rs, 8, 32);
      l[r] = l[r] * alpha[r] + rs;
      m[r] = mn;
      p0[r] = e0;
      p1[r] = e1;
    }
#pragma unroll
    for (int dt = 0; dt < 8; ++dt)
#pragma unroll
      for (int r = 0; r < 8; ++r) o[dt][r] *= alpha[r];

    // ---- P: C-layout -> A-layout via per-wave LDS patch ----
    half_t* pw = Ps + wid * 512;
#pragma unroll
    for (int r = 0; r < 8; ++r) {
      const int Mr = r + hi * 8;
      pw[Mr * 32 + lh]      = (half_t)p0[r];
      pw[Mr * 32 + 16 + lh] = (half_t)p1[r];
    }
    const int ko = hi * 8;
    v16h pf = ldfrag(pw + lh * 32 + ko, pw + lh * 32 + ko + 16);

    // ---- O += P @ V (8 d-tiles, one K=32 step each) ----
#pragma unroll
    for (int dt = 0; dt < 8; ++dt) {
      const half_t* vtp = Vt + (dt * 16 + lh) * 32 + hi * 8;
      v16h vf = ldfrag(vtp, vtp + 16);
      o[dt] = wmma_f16(pf, vf, o[dt]);
    }
    __syncthreads();
  }

  // ---- epilogue: normalize, write f32 attention output ----
#pragma unroll
  for (int r = 0; r < 8; ++r) {
    const float inv = 1.0f / l[r];
    const int row = qrow0 + r + hi * 8;
    float* op = Of + (size_t)row * DMODEL + h * HD + lh;
#pragma unroll
    for (int dt = 0; dt < 8; ++dt) op[dt * 16] = o[dt][r] * inv;
  }
}

// ---------------------------------------------------------------------------
extern "C" void kernel_launch(void* const* d_in, const int* in_sizes, int n_in,
                              void* d_out, int out_size, void* d_ws, size_t ws_size,
                              hipStream_t stream) {
  (void)in_sizes; (void)n_in; (void)out_size; (void)ws_size;
  const float* query = (const float*)d_in[0];
  const float* key_  = (const float*)d_in[1];
  const float* value = (const float*)d_in[2];
  const float* Wq    = (const float*)d_in[3];
  const float* Wk    = (const float*)d_in[4];
  const float* Wv    = (const float*)d_in[5];
  const float* Wo    = (const float*)d_in[6];
  float* out = (float*)d_out;

  char* ws = (char*)d_ws;
  const size_t HBYTES = (size_t)SEQ_ * DMODEL * sizeof(half_t); // 8 MB
  half_t* Qh = (half_t*)(ws);
  half_t* Kh = (half_t*)(ws + HBYTES);
  half_t* Vh = (half_t*)(ws + 2 * HBYTES);
  float*  Af = (float*)(ws + 3 * HBYTES);                       // 16 MB f32

  dim3 gG(DMODEL / 128, SEQ_ / 128), bG(256);
  // QKV projections: f32 in, f16 out
  gemm_nt_kernel<half_t><<<gG, bG, 0, stream>>>(query, Wq, Qh, SEQ_, DMODEL, DMODEL);
  gemm_nt_kernel<half_t><<<gG, bG, 0, stream>>>(key_,  Wk, Kh, SEQ_, DMODEL, DMODEL);
  gemm_nt_kernel<half_t><<<gG, bG, 0, stream>>>(value, Wv, Vh, SEQ_, DMODEL, DMODEL);
  // RoPE on Q and K (in place)
  const int ropeThreads = SEQ_ * HEADS * (HD / 2); // 2,097,152
  rope_kernel<<<ropeThreads / 256, 256, 0, stream>>>(Qh);
  rope_kernel<<<ropeThreads / 256, 256, 0, stream>>>(Kh);
  // Causal flash attention -> f32
  fa_kernel<<<dim3(SEQ_ / 64, HEADS), 128, 0, stream>>>(Qh, Kh, Vh, Af);
  // Output projection: f32 out
  gemm_nt_kernel<float><<<gG, bG, 0, stream>>>(Af, Wo, out, SEQ_, DMODEL, DMODEL);
}
